// PoseGraphRefinementUnit_68891275427959
// MI455X (gfx1250) — matrix operand order repaired and checked by reading, not compiled, not run on hardware
//
#include <hip/hip_runtime.h>
#include <hip/hip_bf16.h>
#include <math.h>

// ---------------------------------------------------------------------------
// Shapes (fixed by the reference): D=128, S=4 (hidden=512), B=2, T=128, N=17
// ---------------------------------------------------------------------------
#define DIM   128
#define HID   512
#define BB    2
#define TT    128
#define NN    17
#define MASK_VAL 1e-28f

typedef __attribute__((ext_vector_type(16))) __bf16 v16bf;
typedef __attribute__((ext_vector_type(8)))  float  v8f;

union AFrag { v16bf v; unsigned u[8]; };

__device__ __forceinline__ unsigned short f32_to_bf16(float f) {
    unsigned u = __float_as_uint(f);
    unsigned r = u + 0x7FFFu + ((u >> 16) & 1u);   // round-to-nearest-even
    return (unsigned short)(r >> 16);
}

__device__ __forceinline__ float gelu_exact(float x) {
    return 0.5f * x * (1.0f + erff(x * 0.70710678118654752f));
}

// ---------------------------------------------------------------------------
// Kernel 1: repack fp32 weights (K x N) into bf16 WMMA B-fragment-major layout.
// Fragment = one wave's B operand for v_wmma_*_16x16x32_bf16: 512 bf16.
// fragIdx = ntile * (K/32) + kstep ; element index within frag = lane*16 + 2v+j
// B layout (ISA 7.12.2): n = ntile*16 + (lane&15); k = kstep*32 + 16*(h>>3) + 8*(lane>>4) + (h&7), h = 2v + j
// ---------------------------------------------------------------------------
__global__ void pack_w_kernel(const float* __restrict__ W,
                              unsigned short* __restrict__ dst,
                              int K, int N) {
    int t = blockIdx.x * blockDim.x + threadIdx.x;
    int total = K * N;
    if (t >= total) return;
    int ksteps = K >> 5;
    int frag   = t >> 9;          // 512 elems per fragment
    int within = t & 511;
    int lane   = within >> 4;
    int h      = within & 15;     // half index: 2v + j
    int kstep  = frag % ksteps;
    int ntile  = frag / ksteps;
    int k = kstep * 32 + ((h >> 3) << 4) + ((lane >> 4) << 3) + (h & 7);
    int n = ntile * 16 + (lane & 15);
    dst[t] = f32_to_bf16(W[(size_t)k * N + n]);
}

// ---------------------------------------------------------------------------
// Kernel 2: sp attention-sum.  out[b,t,i,d] = sum_j softmax_j(sp)[...,j,...]*sp
// One thread per (b,t,i,d).  17 values live in registers.
// ---------------------------------------------------------------------------
__global__ void sp_attn_kernel(const float* __restrict__ sp,
                               float* __restrict__ sp_sum) {
    int id = blockIdx.x * blockDim.x + threadIdx.x;   // B*T*N*D = 557056
    int d  = id & (DIM - 1);
    int i  = (id >> 7) % NN;
    int bt = (id >> 7) / NN;
    size_t base = ((size_t)(bt * NN + i)) * NN * DIM + d;
    float vals[NN];
    float m = -3.0e38f;
#pragma unroll
    for (int j = 0; j < NN; ++j) {
        vals[j] = sp[base + (size_t)j * DIM];
        m = fmaxf(m, vals[j]);
    }
    float s = 0.f;
#pragma unroll
    for (int j = 0; j < NN; ++j) s += expf(vals[j] - m);
    float inv_s = 1.0f / s;
    float acc = 0.f;
#pragma unroll
    for (int j = 0; j < NN; ++j) acc += expf(vals[j] - m) * inv_s * vals[j];
    sp_sum[id] = acc;
}

// ---------------------------------------------------------------------------
// Kernel 3: tv attention-sum with pad/local masks.
// scores = invalid ? 1e-28 : tv ; softmax over t2 ; attn*tv zeroed at invalid.
// Two passes (online max/sum, then weighted sum) to avoid a 3rd 285MB pass.
// ---------------------------------------------------------------------------
__global__ void tv_attn_kernel(const float* __restrict__ tv,
                               const int* __restrict__ pad,
                               const int* __restrict__ lmask,
                               float* __restrict__ tv_sum) {
    int id = blockIdx.x * blockDim.x + threadIdx.x;   // B*N*T*D = 557056
    int d  = id & (DIM - 1);
    int t1 = (id >> 7) & (TT - 1);
    int bn = id >> 14;                 // b*NN + n
    int b  = bn / NN;
    size_t rowoff = (size_t)(bn * TT + t1) * (TT * DIM) + d;
    const int* lmr  = lmask + t1 * TT;
    const int* padr = pad + b * TT;

    float m = -3.0e38f, s = 0.f;
    for (int t2 = 0; t2 < TT; ++t2) {
        float v = tv[rowoff + (size_t)t2 * DIM];
        bool inv = (padr[t2] == 0) || (lmr[t2] == 0);
        float x = inv ? MASK_VAL : v;
        float nm = fmaxf(m, x);
        s = s * expf(m - nm) + expf(x - nm);
        m = nm;
    }
    float inv_s = 1.0f / s;
    float acc = 0.f;
    for (int t2 = 0; t2 < TT; ++t2) {
        float v = tv[rowoff + (size_t)t2 * DIM];
        bool inv = (padr[t2] == 0) || (lmr[t2] == 0);
        if (!inv) acc += expf(v - m) * inv_s * v;
    }
    tv_sum[id] = acc;
}

// ---------------------------------------------------------------------------
// Kernel 4: x = node + (sp_sum@Wsp2n+b, pad-masked) + (tv_sum@Wtv2n+b, transposed)
// then LayerNorm -> xln (fp32).  One 128-thread block per row (b,t,i).
// ---------------------------------------------------------------------------
__global__ __launch_bounds__(128)
void fuse_ln_kernel(const float* __restrict__ node,
                    const float* __restrict__ sp_sum,
                    const float* __restrict__ tv_sum,
                    const int* __restrict__ pad,
                    const float* __restrict__ Wsp2n, const float* __restrict__ bsp2n,
                    const float* __restrict__ Wtv2n, const float* __restrict__ btv2n,
                    const float* __restrict__ ln_g,  const float* __restrict__ ln_b,
                    float* __restrict__ xln) {
    __shared__ float s_sp[DIM];
    __shared__ float s_tv[DIM];
    __shared__ float s_red[DIM];
    int r = blockIdx.x;              // 0..B*T*N-1, ordered (b,t,i)
    int b = r / (TT * NN);
    int rem = r % (TT * NN);
    int t = rem / NN;
    int i = rem % NN;
    int d = threadIdx.x;

    s_sp[d] = sp_sum[(size_t)r * DIM + d];
    s_tv[d] = tv_sum[((size_t)((b * NN + i) * TT + t)) * DIM + d];  // transpose read
    __syncthreads();

    float spv = bsp2n[d], tvv = btv2n[d];
    for (int k = 0; k < DIM; ++k) {
        spv += s_sp[k] * Wsp2n[k * DIM + d];
        tvv += s_tv[k] * Wtv2n[k * DIM + d];
    }
    if (pad[b * TT + t] == 0) spv = 0.f;
    float x = node[(size_t)r * DIM + d] + spv + tvv;

    s_red[d] = x; __syncthreads();
    for (int off = 64; off > 0; off >>= 1) {
        if (d < off) s_red[d] += s_red[d + off];
        __syncthreads();
    }
    float mu = s_red[0] * (1.0f / DIM);
    __syncthreads();
    float xc = x - mu;
    s_red[d] = xc * xc; __syncthreads();
    for (int off = 64; off > 0; off >>= 1) {
        if (d < off) s_red[d] += s_red[d + off];
        __syncthreads();
    }
    float var = s_red[0] * (1.0f / DIM);
    xln[(size_t)r * DIM + d] = xc * rsqrtf(var + 1e-5f) * ln_g[d] + ln_b[d];
}

// ---------------------------------------------------------------------------
// Kernel 5: fused WMMA MLP  out = gelu(X@W1+b1)@W2+b2  (128 -> 512 -> 128)
// Block = 128 threads (4 waves), M-tile = 32 rows.
//  wave (w>>1) -> row half,  wave (w&1) -> N-tile parity (splits N tiles).
// A fragments gathered from LDS per ISA 16-bit A layout; B fragments loaded
// pre-packed (coalesced b128 loads).  maskMode==1 applies local_mask (tv_out).
// ---------------------------------------------------------------------------
__global__ __launch_bounds__(128)
void mlp_wmma_kernel(const float* __restrict__ X,
                     const unsigned short* __restrict__ W1p, const float* __restrict__ b1,
                     const unsigned short* __restrict__ W2p, const float* __restrict__ b2,
                     float* __restrict__ Out,
                     const int* __restrict__ lmask, int maskMode) {
    __shared__ unsigned short Xs[32][DIM + 4];   // bf16 X tile, padded
    __shared__ unsigned short Hs[32][HID + 8];   // bf16 hidden tile, padded

    const int tid  = threadIdx.x;
    const int wave = tid >> 5;
    const int lane = tid & 31;
    const long rowBlock = (long)blockIdx.x * 32;

    // --- stage X tile (32 rows x 128 fp32) -> bf16 LDS, coalesced float4 ---
    const float4* Xv = (const float4*)(X + (size_t)rowBlock * DIM);
#pragma unroll
    for (int it = 0; it < 8; ++it) {
        int idx = tid + it * 128;          // 1024 float4 total
        float4 f = Xv[idx];
        int r = idx >> 5, c = (idx & 31) << 2;
        Xs[r][c + 0] = f32_to_bf16(f.x);
        Xs[r][c + 1] = f32_to_bf16(f.y);
        Xs[r][c + 2] = f32_to_bf16(f.z);
        Xs[r][c + 3] = f32_to_bf16(f.w);
    }
    __syncthreads();

    const int rowHalf = wave >> 1;               // 0/1 -> rows [0..15] / [16..31]
    const int par     = wave & 1;                // N-tile parity
    const int mrow    = rowHalf * 16 + (lane & 15);
    const int ncl     = lane & 15;               // output column within tile
    const int laneHi  = lane >> 4;

    // --- A fragments for GEMM1 (K=128 -> 4 k-steps), loaded once ---
    AFrag a1[4];
#pragma unroll
    for (int ks = 0; ks < 4; ++ks)
#pragma unroll
        for (int v = 0; v < 8; ++v) {
            int k = ks * 32 + ((v >> 2) << 4) + (laneHi << 3) + ((v & 3) << 1);
            a1[ks].u[v] = *(const unsigned*)&Xs[mrow][k];
        }

    // --- GEMM1 + bias + GELU -> Hs (bf16) ---
    for (int nt = par; nt < HID / 16; nt += 2) {
        v8f acc = {0.f, 0.f, 0.f, 0.f, 0.f, 0.f, 0.f, 0.f};
#pragma unroll
        for (int ks = 0; ks < 4; ++ks) {
            AFrag bf;
            const unsigned* wp = (const unsigned*)W1p + ((size_t)(nt * 4 + ks)) * 256 + lane * 8;
            __builtin_prefetch(wp + 512, 0, 1);
#pragma unroll
            for (int v = 0; v < 8; ++v) bf.u[v] = wp[v];
            acc = __builtin_amdgcn_wmma_f32_16x16x32_bf16(
                false, a1[ks].v, false, bf.v, (short)0, acc, false, false);
        }
        float bias = b1[nt * 16 + ncl];
        int   ncol = nt * 16 + ncl;
#pragma unroll
        for (int r = 0; r < 8; ++r) {
            float h = gelu_exact(acc[r] + bias);
            int m = rowHalf * 16 + r + (laneHi << 3);
            Hs[m][ncol] = f32_to_bf16(h);
        }
    }
    __syncthreads();

    // --- GEMM2: Out = H @ W2 + b2 (K=512 -> 16 k-steps, 8 N tiles) ---
    for (int nt = par; nt < DIM / 16; nt += 2) {
        v8f acc = {0.f, 0.f, 0.f, 0.f, 0.f, 0.f, 0.f, 0.f};
#pragma unroll
        for (int ks = 0; ks < 16; ++ks) {
            AFrag af, bf;
#pragma unroll
            for (int v = 0; v < 8; ++v) {
                int k = ks * 32 + ((v >> 2) << 4) + (laneHi << 3) + ((v & 3) << 1);
                af.u[v] = *(const unsigned*)&Hs[mrow][k];
            }
            const unsigned* wp = (const unsigned*)W2p + ((size_t)(nt * 16 + ks)) * 256 + lane * 8;
#pragma unroll
            for (int v = 0; v < 8; ++v) bf.u[v] = wp[v];
            acc = __builtin_amdgcn_wmma_f32_16x16x32_bf16(
                false, af.v, false, bf.v, (short)0, acc, false, false);
        }
        float bias = b2[nt * 16 + ncl];
        int   ncol = nt * 16 + ncl;
#pragma unroll
        for (int r = 0; r < 8; ++r) {
            int  m   = rowHalf * 16 + r + (laneHi << 3);
            long row = rowBlock + m;
            float o = acc[r] + bias;
            if (maskMode == 1) {
                int t2 = (int)(row & (TT - 1));
                int t1 = (int)((row >> 7) & (TT - 1));
                if (lmask[t1 * TT + t2] == 0) o = 0.f;
            }
            Out[row * DIM + ncol] = o;
        }
    }
}

// ---------------------------------------------------------------------------
// Host-side launcher
// ---------------------------------------------------------------------------
extern "C" void kernel_launch(void* const* d_in, const int* in_sizes, int n_in,
                              void* d_out, int out_size, void* d_ws, size_t ws_size,
                              hipStream_t stream) {
    // Inputs (setup_inputs order)
    const float* node  = (const float*)d_in[0];
    const float* sp    = (const float*)d_in[1];
    const float* tv    = (const float*)d_in[2];
    const int*   pad   = (const int*)  d_in[3];   // (B,1,T)
    const int*   lmask = (const int*)  d_in[4];   // (T,T)
    const float* ln_g  = (const float*)d_in[5];
    const float* ln_b  = (const float*)d_in[6];
    const float* Wn1   = (const float*)d_in[7];
    const float* bn1   = (const float*)d_in[8];
    const float* Wn2   = (const float*)d_in[9];
    const float* bn2   = (const float*)d_in[10];
    const float* Wsp1  = (const float*)d_in[11];
    const float* bsp1  = (const float*)d_in[12];
    const float* Wsp2  = (const float*)d_in[13];
    const float* bsp2  = (const float*)d_in[14];
    const float* Wtv1  = (const float*)d_in[15];
    const float* btv1  = (const float*)d_in[16];
    const float* Wtv2  = (const float*)d_in[17];
    const float* btv2  = (const float*)d_in[18];
    const float* Wsp2n = (const float*)d_in[19];
    const float* bsp2n = (const float*)d_in[20];
    const float* Wtv2n = (const float*)d_in[21];
    const float* btv2n = (const float*)d_in[22];

    float* out = (float*)d_out;
    const long NODE_ROWS = (long)BB * TT * NN;              // 4352
    const long SP_ROWS   = (long)BB * TT * NN * NN;         // 73984
    const long TV_ROWS   = (long)BB * NN * TT * TT;         // 557056
    float* node_out = out;                                  // 557,056
    float* sp_out   = out + NODE_ROWS * DIM;                // +9,469,952
    float* tv_out   = out + NODE_ROWS * DIM + SP_ROWS * DIM;

    // Workspace layout (bytes)
    char* ws = (char*)d_ws;
    const size_t WSZ = (size_t)DIM * HID * sizeof(unsigned short);  // 131072
    unsigned short* Wn1p  = (unsigned short*)(ws + 0 * WSZ);
    unsigned short* Wn2p  = (unsigned short*)(ws + 1 * WSZ);
    unsigned short* Wsp1p = (unsigned short*)(ws + 2 * WSZ);
    unsigned short* Wsp2p = (unsigned short*)(ws + 3 * WSZ);
    unsigned short* Wtv1p = (unsigned short*)(ws + 4 * WSZ);
    unsigned short* Wtv2p = (unsigned short*)(ws + 5 * WSZ);
    size_t off = 6 * WSZ;
    float* sp_sum = (float*)(ws + off);             off += (size_t)NODE_ROWS * DIM * 4;
    float* tv_sum = (float*)(ws + off);             off += (size_t)NODE_ROWS * DIM * 4;
    float* xln    = (float*)(ws + off);             // + NODE_ROWS*DIM*4

    // 1) pack weights -> bf16 fragment-major
    {
        int total = DIM * HID;                    // 65536 per matrix
        dim3 g((total + 255) / 256), blk(256);
        pack_w_kernel<<<g, blk, 0, stream>>>(Wn1,  Wn1p,  DIM, HID);
        pack_w_kernel<<<g, blk, 0, stream>>>(Wn2,  Wn2p,  HID, DIM);
        pack_w_kernel<<<g, blk, 0, stream>>>(Wsp1, Wsp1p, DIM, HID);
        pack_w_kernel<<<g, blk, 0, stream>>>(Wsp2, Wsp2p, HID, DIM);
        pack_w_kernel<<<g, blk, 0, stream>>>(Wtv1, Wtv1p, DIM, HID);
        pack_w_kernel<<<g, blk, 0, stream>>>(Wtv2, Wtv2p, HID, DIM);
    }

    // 2) attention sums
    sp_attn_kernel<<<dim3((unsigned)(NODE_ROWS * DIM / 256)), dim3(256), 0, stream>>>(sp, sp_sum);
    tv_attn_kernel<<<dim3((unsigned)(NODE_ROWS * DIM / 256)), dim3(256), 0, stream>>>(tv, pad, lmask, tv_sum);

    // 3) residual fuse + layernorm
    fuse_ln_kernel<<<dim3((unsigned)NODE_ROWS), dim3(128), 0, stream>>>(
        node, sp_sum, tv_sum, pad, Wsp2n, bsp2n, Wtv2n, btv2n, ln_g, ln_b, xln);

    // 4) the three WMMA MLPs
    mlp_wmma_kernel<<<dim3((unsigned)(NODE_ROWS / 32)), dim3(128), 0, stream>>>(
        xln, Wn1p, bn1, Wn2p, bn2, node_out, lmask, 0);
    mlp_wmma_kernel<<<dim3((unsigned)(SP_ROWS / 32)), dim3(128), 0, stream>>>(
        sp, Wsp1p, bsp1, Wsp2p, bsp2, sp_out, lmask, 0);
    mlp_wmma_kernel<<<dim3((unsigned)(TV_ROWS / 32)), dim3(128), 0, stream>>>(
        tv, Wtv1p, btv1, Wtv2p, btv2, tv_out, lmask, 1);
}
